// GATmodel_68676527063540
// MI455X (gfx1250) — matrix operand order repaired and checked by reading, not compiled, not run on hardware
//
#include <hip/hip_runtime.h>
#include <hip/hip_bf16.h>
#include <math.h>
#include <stdint.h>

// ---------------------------------------------------------------------------
// Problem constants (match reference)
// ---------------------------------------------------------------------------
#define NN      10000
#define EE      100000
#define TT      4
#define ROWS    (NN * TT)        // 40000 node-time rows
#define HFE     64               // per-head feature width (both layers)
#define NH      4                // heads (both layers)
#define OC      256              // NH * HFE, output channels of both GAT fc's
#define K1      64               // K of GEMM1 (h -> fcW1)
#define K2      256              // K of GEMM2 (act1 -> fcW2)

typedef _Float16 v16h __attribute__((ext_vector_type(16)));
typedef _Float16 v8h  __attribute__((ext_vector_type(8)));
typedef float    v8f  __attribute__((ext_vector_type(8)));
typedef unsigned int u32x4 __attribute__((ext_vector_type(4)));
typedef int          i32x4 __attribute__((ext_vector_type(4)));
typedef int          i32x8 __attribute__((ext_vector_type(8)));

__device__ __forceinline__ float sigmoidf_(float x) {
    return 1.0f / (1.0f + __expf(-x));
}

// ---------------------------------------------------------------------------
// fp32 -> fp16 weight conversion
// ---------------------------------------------------------------------------
__global__ void k_cvt_f16(const float* __restrict__ in, _Float16* __restrict__ out, int n) {
    int i = blockIdx.x * 256 + threadIdx.x;
    if (i < n) out[i] = (_Float16)in[i];
}

// ---------------------------------------------------------------------------
// MLP layer 1: hidden = relu(x @ W1^T + b1), x[row][k] = feat[n][k][t]
// ---------------------------------------------------------------------------
__global__ void k_mlp1(const float* __restrict__ feat, const float* __restrict__ W1,
                       const float* __restrict__ b1, float* __restrict__ hidden) {
    int idx = blockIdx.x * 256 + threadIdx.x;   // ROWS*64
    int row = idx >> 6, c = idx & 63;
    int n = row >> 2, t = row & 3;
    float s = b1[c];
#pragma unroll
    for (int k = 0; k < 16; ++k)
        s += feat[(n * 16 + k) * 4 + t] * W1[c * 16 + k];
    hidden[idx] = s > 0.f ? s : 0.f;
}

// ---------------------------------------------------------------------------
// MLP layer 2: h = hidden @ W2^T + b2 ; emit fp16 for the WMMA GEMM
// ---------------------------------------------------------------------------
__global__ void k_mlp2(const float* __restrict__ hidden, const float* __restrict__ W2,
                       const float* __restrict__ b2, _Float16* __restrict__ hf) {
    int idx = blockIdx.x * 256 + threadIdx.x;   // ROWS*64
    int row = idx >> 6, c = idx & 63;
    float s = b2[c];
    const float* hp = hidden + (size_t)row * 64;
#pragma unroll 8
    for (int k = 0; k < 64; ++k) s += hp[k] * W2[c * 64 + k];
    hf[idx] = (_Float16)s;
}

// ---------------------------------------------------------------------------
// WMMA GEMM: C[ROWS x 256] = A[ROWS x K] @ B[256 x K]^T
// Block = 128 threads (4 waves). Block computes a 64(M) x 64(N) output panel:
// wave w owns M-tile (blockIdx.y*4+w), and iterates 4 N-tiles.
// The 64 x K fp16 weight panel is DMA'd into LDS once per block with the
// Tensor Data Mover (tensor_load_to_lds, 2D tile, data_size=2B), then read
// back with ds_load_b128. A is read direct from global and register-reused
// across the 4 N-tiles. K consumed in chunks of 32 via v_wmma_f32_16x16x32_f16.
//   A 16x32 tile (16-bit A): lanes 0-15 -> M=lane, K = kc+[0..7], kc+16+[0..7];
//   lanes 16-31 -> M=lane-16, K = kc+8+[0..7], kc+24+[0..7].
//   B 32x16 tile: lanes 0-15 -> N=lane, K = kc+[0..7], kc+16+[0..7]; lanes 16-31 -> N=lane-16,
//   K = kc+8+[0..7], kc+24+[0..7]. fcW row o = B column o, so two 8-half runs of the
//   staged panel row are one lane's payload.
// ---------------------------------------------------------------------------
__global__ void __launch_bounds__(128)
k_gemm_wmma(const _Float16* __restrict__ A, const _Float16* __restrict__ B,
            float* __restrict__ C, int K) {
    __shared__ __align__(16) _Float16 Bsh[64 * 256];   // 32 KB max (K<=256)

    const int tid   = threadIdx.x;
    const int wave  = tid >> 5;
    const int lane  = tid & 31;
    const bool hi   = lane >= 16;
    const int l15   = lane & 15;
    const int tileM  = (blockIdx.y * 4 + wave) * 16;
    const int tileN0 = blockIdx.x * 64;

    // ---- TDM: stage B panel [tileN0 .. tileN0+63] x K halves into LDS -----
    if (tid < 32) {
        unsigned long long gaddr =
            (unsigned long long)(uintptr_t)(B + (size_t)tileN0 * K);
        unsigned int ldsa = (unsigned int)(uintptr_t)&Bsh[0]; // low 32b = LDS offset
        unsigned int Ku = (unsigned int)K;

        u32x4 g0;
        g0[0] = 1u;                                          // count=1 (user D#)
        g0[1] = ldsa;                                        // lds_addr
        g0[2] = (unsigned int)(gaddr & 0xffffffffu);         // global_addr[31:0]
        g0[3] = (unsigned int)((gaddr >> 32) & 0x01ffffffu)  // global_addr[56:32]
              | 0x80000000u;                                 // type=2 ("image")

        i32x8 g1;
        g1[0] = 0x00010000;                                  // data_size=1 -> 2 bytes
        g1[1] = (int)((Ku & 0xffffu) << 16);                 // tensor_dim0[15:0]
        g1[2] = (int)(((Ku >> 16) & 0xffffu) | (64u << 16)); // td0[31:16] | tensor_dim1[15:0]=64
        g1[3] = (int)((Ku & 0xffffu) << 16);                 // td1[31:16]=0 | tile_dim0=K
        g1[4] = 64;                                          // tile_dim1=64, tile_dim2=0
        g1[5] = (int)Ku;                                     // tensor_dim0_stride[31:0]=K
        g1[6] = 0;                                           // stride hi / dim1_stride lo
        g1[7] = 0;

        i32x4 z4 = {0, 0, 0, 0};
#if defined(__clang_major__) && (__clang_major__ >= 23)
        i32x8 z8 = {0, 0, 0, 0, 0, 0, 0, 0};
        __builtin_amdgcn_tensor_load_to_lds(g0, g1, z4, z4, z8, 0);
#else
        __builtin_amdgcn_tensor_load_to_lds(g0, g1, z4, z4, 0);
#endif
        __builtin_amdgcn_s_wait_tensorcnt(0);
    }
    __syncthreads();

    // ---- WMMA main loop ----------------------------------------------------
    const size_t aRow = (size_t)(tileM + l15) * (size_t)K;
    v8f acc[4] = {{}, {}, {}, {}};

    for (int kc = 0; kc < K; kc += 32) {
        const _Float16* ap = A + aRow + kc + (hi ? 8 : 0);
        v8h a_lo = *(const v8h*)(ap);
        v8h a_hi = *(const v8h*)(ap + 16);
        v16h av;
#pragma unroll
        for (int i = 0; i < 8; ++i) { av[i] = a_lo[i]; av[i + 8] = a_hi[i]; }

#pragma unroll
        for (int nt = 0; nt < 4; ++nt) {
            const _Float16* bp = &Bsh[(size_t)(nt * 16 + l15) * K + kc + (hi ? 8 : 0)];
            v8h b_lo = *(const v8h*)(bp);
            v8h b_hi = *(const v8h*)(bp + 16);
            v16h bv;
#pragma unroll
            for (int i = 0; i < 8; ++i) { bv[i] = b_lo[i]; bv[i + 8] = b_hi[i]; }
            acc[nt] = __builtin_amdgcn_wmma_f32_16x16x32_f16(
                false, av, false, bv, (short)0, acc[nt], false, false);
        }
    }

    // D layout: VGPR r -> (M = tileM + r + (hi?8:0), N = tileN + l15)
#pragma unroll
    for (int nt = 0; nt < 4; ++nt) {
        float* cp = C + (size_t)(tileM + (hi ? 8 : 0)) * OC + tileN0 + nt * 16 + l15;
#pragma unroll
        for (int r = 0; r < 8; ++r) cp[(size_t)r * OC] = acc[nt][r];
    }
}

// ---------------------------------------------------------------------------
// Attention dots: el[row,h] = sum_f ft[row,h,f]*al[h,f]  (same for er/ar)
// One wave per (row, head); F=64 -> 2 elems/lane then wave reduction.
// ---------------------------------------------------------------------------
__global__ void k_attn(const float* __restrict__ ft, const float* __restrict__ al,
                       const float* __restrict__ ar, float* __restrict__ el,
                       float* __restrict__ er) {
    int wave = threadIdx.x >> 5;
    int lane = threadIdx.x & 31;
    int task = blockIdx.x * 8 + wave;          // < ROWS*NH
    int row  = task >> 2;
    int h    = task & 3;
    const float* fp = ft + (size_t)row * OC + h * HFE;
    float v0 = fp[lane], v1 = fp[lane + 32];
    float sl = v0 * al[h * HFE + lane] + v1 * al[h * HFE + lane + 32];
    float sr = v0 * ar[h * HFE + lane] + v1 * ar[h * HFE + lane + 32];
#pragma unroll
    for (int off = 16; off; off >>= 1) {
        sl += __shfl_xor(sl, off, 32);
        sr += __shfl_xor(sr, off, 32);
    }
    if (lane == 0) { el[task] = sl; er[task] = sr; }
}

// ---------------------------------------------------------------------------
// Edge pass A: ex = exp(leaky_relu(el[src]+er[dst])), denom[dst] += ex
// (max-shift omitted: mathematically identical softmax, exponents are O(1))
// ---------------------------------------------------------------------------
__global__ void k_edgeA(const int* __restrict__ src, const int* __restrict__ dst,
                        const float* __restrict__ el, const float* __restrict__ er,
                        float* __restrict__ ex, float* __restrict__ denom) {
    int idx = blockIdx.x * 256 + threadIdx.x;  // < E*T*NH
    int e  = idx >> 4;
    int th = idx & 15;
    int t = th >> 2, h = th & 3;
    int s = src[e], d = dst[e];
    float v = el[(s * 4 + t) * 4 + h] + er[(d * 4 + t) * 4 + h];
    v = v > 0.f ? v : 0.2f * v;
    float xv = __expf(v);
    ex[idx] = xv;
    atomicAdd(&denom[(d * 4 + t) * 4 + h], xv);
}

// ---------------------------------------------------------------------------
// Edge pass B: agg[dst] += ft[src] * (ex / denom[dst]); one block per (e,t)
// ---------------------------------------------------------------------------
__global__ void k_edgeB(const int* __restrict__ src, const int* __restrict__ dst,
                        const float* __restrict__ ft, const float* __restrict__ ex,
                        const float* __restrict__ denom, float* __restrict__ agg) {
    int b = blockIdx.x;                        // e*T + t
    int e = b >> 2, t = b & 3;
    int c = threadIdx.x;                       // 0..255
    int h = c >> 6;
    int s = src[e], d = dst[e];
    float a = ex[b * 4 + h] / denom[(d * 4 + t) * 4 + h];
    float val = ft[(size_t)(s * 4 + t) * OC + c] * a;
    atomicAdd(&agg[(size_t)(d * 4 + t) * OC + c], val);
}

// ---------------------------------------------------------------------------
// act1 = (fp16) sigmoid(agg + bias)
// ---------------------------------------------------------------------------
__global__ void k_post1(const float* __restrict__ agg, const float* __restrict__ bias,
                        _Float16* __restrict__ act) {
    int idx = blockIdx.x * 256 + threadIdx.x;  // ROWS*OC
    act[idx] = (_Float16)sigmoidf_(agg[idx] + bias[idx & 255]);
}

// ---------------------------------------------------------------------------
// out = sigmoid(out + bias)  (in place on d_out after the atomic aggregation)
// ---------------------------------------------------------------------------
__global__ void k_final(float* __restrict__ out, const float* __restrict__ bias) {
    int idx = blockIdx.x * 256 + threadIdx.x;  // ROWS*OC
    out[idx] = sigmoidf_(out[idx] + bias[idx & 255]);
}

// ---------------------------------------------------------------------------
// Launch
// ---------------------------------------------------------------------------
extern "C" void kernel_launch(void* const* d_in, const int* in_sizes, int n_in,
                              void* d_out, int out_size, void* d_ws, size_t ws_size,
                              hipStream_t stream) {
    const float* feat  = (const float*)d_in[0];
    const int*   src   = (const int*)  d_in[1];
    const int*   dst   = (const int*)  d_in[2];
    const float* W1    = (const float*)d_in[3];
    const float* b1    = (const float*)d_in[4];
    const float* W2    = (const float*)d_in[5];
    const float* b2    = (const float*)d_in[6];
    const float* fcW1  = (const float*)d_in[7];
    const float* al1   = (const float*)d_in[8];
    const float* ar1   = (const float*)d_in[9];
    const float* bias1 = (const float*)d_in[10];
    const float* fcW2  = (const float*)d_in[11];
    const float* al2   = (const float*)d_in[12];
    const float* ar2   = (const float*)d_in[13];
    const float* bias2 = (const float*)d_in[14];
    float* out = (float*)d_out;

    // ---- workspace carve-up (256B aligned) --------------------------------
    char* ws = (char*)d_ws;
    size_t off = 0;
    auto carve = [&](size_t bytes) { char* p = ws + off; off += (bytes + 255) & ~(size_t)255; return p; };
    float*    hidden = (float*)   carve((size_t)ROWS * 64 * 4);   // 10.24 MB
    _Float16* h16    = (_Float16*)carve((size_t)ROWS * 64 * 2);   //  5.12 MB
    float*    ft     = (float*)   carve((size_t)ROWS * OC * 4);   // 40.96 MB (reused layer2)
    float*    agg1   = (float*)   carve((size_t)ROWS * OC * 4);   // 40.96 MB
    _Float16* act16  = (_Float16*)carve((size_t)ROWS * OC * 2);   // 20.48 MB
    float*    el     = (float*)   carve((size_t)ROWS * NH * 4);   // 0.64 MB (reused)
    float*    er     = (float*)   carve((size_t)ROWS * NH * 4);
    float*    denom  = (float*)   carve((size_t)ROWS * NH * 4);
    float*    ex     = (float*)   carve((size_t)EE * TT * NH * 4);// 6.4 MB (reused)
    _Float16* fcW1h  = (_Float16*)carve((size_t)OC * K1 * 2);
    _Float16* fcW2h  = (_Float16*)carve((size_t)OC * K2 * 2);

    // ---- weight conversion ------------------------------------------------
    k_cvt_f16<<<(OC * K1 + 255) / 256, 256, 0, stream>>>(fcW1, fcW1h, OC * K1);
    k_cvt_f16<<<(OC * K2 + 255) / 256, 256, 0, stream>>>(fcW2, fcW2h, OC * K2);

    // ---- node MLP ---------------------------------------------------------
    k_mlp1<<<ROWS * 64 / 256, 256, 0, stream>>>(feat, W1, b1, hidden);
    k_mlp2<<<ROWS * 64 / 256, 256, 0, stream>>>(hidden, W2, b2, h16);

    // ---- GAT layer 1 ------------------------------------------------------
    {
        dim3 g(OC / 64, ROWS / 64);   // 4 x 625 blocks, 128 threads each
        k_gemm_wmma<<<g, 128, 0, stream>>>(h16, fcW1h, ft, K1);
    }
    k_attn<<<ROWS * NH / 8, 256, 0, stream>>>(ft, al1, ar1, el, er);
    hipMemsetAsync(denom, 0, (size_t)ROWS * NH * 4, stream);
    hipMemsetAsync(agg1, 0, (size_t)ROWS * OC * 4, stream);
    k_edgeA<<<EE * TT * NH / 256, 256, 0, stream>>>(src, dst, el, er, ex, denom);
    k_edgeB<<<EE * TT, 256, 0, stream>>>(src, dst, ft, ex, denom, agg1);
    k_post1<<<ROWS * OC / 256, 256, 0, stream>>>(agg1, bias1, act16);

    // ---- GAT layer 2 (buffers reused; aggregation lands in d_out) ---------
    {
        dim3 g(OC / 64, ROWS / 64);
        k_gemm_wmma<<<g, 128, 0, stream>>>(act16, fcW2h, ft, K2);
    }
    k_attn<<<ROWS * NH / 8, 256, 0, stream>>>(ft, al2, ar2, el, er);
    hipMemsetAsync(denom, 0, (size_t)ROWS * NH * 4, stream);
    hipMemsetAsync(out, 0, (size_t)ROWS * OC * 4, stream);
    k_edgeA<<<EE * TT * NH / 256, 256, 0, stream>>>(src, dst, el, er, ex, denom);
    k_edgeB<<<EE * TT, 256, 0, stream>>>(src, dst, ft, ex, denom, out);
    k_final<<<ROWS * OC / 256, 256, 0, stream>>>(out, bias2);
}
